// EnhancedJointECGModel_23691039605278
// MI455X (gfx1250) — matrix operand order repaired and checked by reading, not compiled, not run on hardware
//
#include <hip/hip_runtime.h>
#include <cstdint>
#include <cstddef>

// ---------------------------------------------------------------------------
// Types for CDNA5 WMMA (gfx1250, wave32)
// ---------------------------------------------------------------------------
typedef __attribute__((ext_vector_type(16))) __bf16 v16bf;   // 32 B
typedef __attribute__((ext_vector_type(8)))  __bf16 v8bf;    // 16 B
typedef __attribute__((ext_vector_type(8)))  float  v8f;

__device__ __forceinline__ unsigned short f2bf(float x) {
  unsigned int u = __float_as_uint(x);
  u += 0x7FFFu + ((u >> 16) & 1u);          // round-to-nearest-even
  return (unsigned short)(u >> 16);
}

// ---------------------------------------------------------------------------
// Utility kernels
// ---------------------------------------------------------------------------
__global__ void k_zero(float* p, long n) {
  long i = (long)blockIdx.x * blockDim.x + threadIdx.x;
  long s = (long)gridDim.x * blockDim.x;
  for (; i < n; i += s) p[i] = 0.f;
}

__global__ void k_fill(float* p, float v, long n) {
  long i = (long)blockIdx.x * blockDim.x + threadIdx.x;
  long s = (long)gridDim.x * blockDim.x;
  for (; i < n; i += s) p[i] = v;
}

__global__ void k_f32_to_bf16(const float* __restrict__ x, unsigned short* __restrict__ y, long n) {
  long i = (long)blockIdx.x * blockDim.x + threadIdx.x;
  long s = (long)gridDim.x * blockDim.x;
  for (; i < n; i += s) y[i] = f2bf(x[i]);
}

// ---------------------------------------------------------------------------
// bf16 WMMA GEMM:  C[M,Nn] = A[M,K](bf16,row-major) x W[Nn,K](bf16,row-major)^T
// Key point: with row-major K-contiguous operands, each lane's WMMA fragment
// is made of contiguous, 16B/32B-aligned runs -> load fragments DIRECTLY from
// global memory as b128 vectors. No LDS, no barriers, no scalarized staging.
//   A 16x32 frag (lane g,r): row r, K run [16g,16g+16)
//   B 32x16 frag (lane g,r): W row n0+r, K run [16g,16g+16)
// Each wave computes a 16(M) x 64(N) strip: A frag reused across 4 WMMAs.
// Block = 256 threads = 8 waves -> block tile 128(M) x 64(N).
// Requires: K % 32 == 0, Nn % 64 == 0, M % 16 == 0 (all true here).
// ---------------------------------------------------------------------------
__global__ void k_gemm_bf16(const unsigned short* __restrict__ A,
                            const unsigned short* __restrict__ W,
                            const float* __restrict__ bias,
                            float* __restrict__ C,
                            unsigned short* __restrict__ Cbf,
                            int M, int Nn, int K, int relu)
{
  const int wave = threadIdx.x >> 5;
  const int lane = threadIdx.x & 31;
  const int m0 = blockIdx.x * 128;
  const int n0 = blockIdx.y * 64;
  const int g = lane >> 4;       // lane half
  const int r = lane & 15;

  const int wm = m0 + wave * 16;
  if (wm >= M) return;           // wave-uniform: EXEC stays all-ones for WMMA

  union Frag { v16bf v; v8bf h[2]; };
  union Acc  { v8f v; float f[8]; };
  Acc acc[4];
#pragma unroll
  for (int nt = 0; nt < 4; ++nt)
#pragma unroll
    for (int j = 0; j < 8; ++j) acc[nt].f[j] = 0.f;

  const unsigned short* aRow = A + (size_t)(wm + r) * K;
  for (int k0 = 0; k0 < K; k0 += 32) {
    Frag fa;
    fa.v = *(const v16bf*)(aRow + k0 + 16 * g);         // 32B aligned
#pragma unroll
    for (int nt = 0; nt < 4; ++nt) {
      Frag fb;
      fb.v = *(const v16bf*)(W + (size_t)(n0 + nt * 16 + r) * K + k0 + 16 * g); // 32B aligned
      acc[nt].v = __builtin_amdgcn_wmma_f32_16x16x32_bf16(false, fa.v, false, fb.v,
                                                          (short)0, acc[nt].v, false, false);
    }
  }

#pragma unroll
  for (int nt = 0; nt < 4; ++nt) {
    int col = n0 + nt * 16 + r;
    float bv = bias ? bias[col] : 0.f;
#pragma unroll
    for (int j = 0; j < 8; ++j) {
      int row = wm + j + 8 * g;  // C layout: VGPR j -> M = j + 8*(lane>=16)
      float v = acc[nt].f[j] + bv;
      if (relu) v = fmaxf(v, 0.f);
      C[(size_t)row * Nn + col] = v;
      if (Cbf) Cbf[(size_t)row * Nn + col] = f2bf(v);
    }
  }
}

// ---------------------------------------------------------------------------
// Encoder conv kernels (K=5, pad=2)
// ---------------------------------------------------------------------------
__global__ void k_conv1(const float* __restrict__ x, const float* __restrict__ w,
                        const float* __restrict__ b, float* __restrict__ y, int N) {
  const int L = 187;
  long total = (long)N * 32 * L;
  long i = (long)blockIdx.x * blockDim.x + threadIdx.x;
  long s = (long)gridDim.x * blockDim.x;
  for (; i < total; i += s) {
    int t = (int)(i % L);
    long rest = i / L;
    int c = (int)(rest % 32);
    long n = rest / 32;
    float acc = b[c];
#pragma unroll
    for (int k = 0; k < 5; ++k) {
      int p = t + k - 2;
      if (p >= 0 && p < L) acc += x[n * L + p] * w[c * 5 + k];
    }
    y[i] = acc;
  }
}

// One sample per block; input tile kept LDS-resident (<= 2976 floats).
__global__ void k_conv1d(const float* __restrict__ x, const float* __restrict__ w,
                         const float* __restrict__ b, float* __restrict__ y,
                         int Cin, int Cout, int L) {
  __shared__ float lIn[2976];
  int n = blockIdx.x;
  int nin = Cin * L;
  for (int i = threadIdx.x; i < nin; i += blockDim.x)
    lIn[i] = x[(size_t)n * nin + i];
  __syncthreads();
  int nout = Cout * L;
  for (int i = threadIdx.x; i < nout; i += blockDim.x) {
    int o = i / L, t = i % L;
    float acc = b[o];
    const float* wo = w + (size_t)o * Cin * 5;
    for (int c = 0; c < Cin; ++c) {
      const float* lc = lIn + c * L;
      const float* wc = wo + c * 5;
#pragma unroll
      for (int k = 0; k < 5; ++k) {
        int p = t + k - 2;
        if (p >= 0 && p < L) acc += lc[p] * wc[k];
      }
    }
    y[(size_t)n * nout + i] = acc;
  }
}

// ---------------------------------------------------------------------------
// BatchNorm (training-mode batch stats) helpers
// ---------------------------------------------------------------------------
__global__ void k_bn_stats(const float* __restrict__ x, float* __restrict__ stats,
                           long total, int C, int L) {
  __shared__ float ssum[128];
  __shared__ float ssq[128];
  for (int c = threadIdx.x; c < C; c += blockDim.x) { ssum[c] = 0.f; ssq[c] = 0.f; }
  __syncthreads();
  long i = (long)blockIdx.x * blockDim.x + threadIdx.x;
  long s = (long)gridDim.x * blockDim.x;
  for (; i < total; i += s) {
    float v = x[i];
    int c = (int)((i / L) % C);
    atomicAdd(&ssum[c], v);
    atomicAdd(&ssq[c], v * v);
  }
  __syncthreads();
  for (int c = threadIdx.x; c < C; c += blockDim.x) {
    atomicAdd(&stats[c], ssum[c]);
    atomicAdd(&stats[C + c], ssq[c]);
  }
}

// fused BN(batch stats) + ReLU + maxpool2
__global__ void k_bn_relu_pool(const float* __restrict__ x, const float* __restrict__ stats,
                               const float* __restrict__ gam, const float* __restrict__ bet,
                               float* __restrict__ y, long totalOut,
                               int C, int L, int Lout, float invcnt) {
  long i = (long)blockIdx.x * blockDim.x + threadIdx.x;
  long s = (long)gridDim.x * blockDim.x;
  for (; i < totalOut; i += s) {
    int t = (int)(i % Lout);
    long rest = i / Lout;
    int c = (int)(rest % C);
    long n = rest / C;
    float m = stats[c] * invcnt;
    float var = stats[C + c] * invcnt - m * m;
    float sc = gam[c] * rsqrtf(var + 1e-5f);
    float sh = bet[c] - m * sc;
    const float* xp = x + ((size_t)n * C + c) * L + 2 * t;
    float v0 = fmaxf(xp[0] * sc + sh, 0.f);
    float v1 = fmaxf(xp[1] * sc + sh, 0.f);
    y[i] = fmaxf(v0, v1);
  }
}

// fused BN + ReLU (decoder)
__global__ void k_bn_act(const float* __restrict__ x, const float* __restrict__ stats,
                         const float* __restrict__ gam, const float* __restrict__ bet,
                         float* __restrict__ y, long total, int C, int L, float invcnt) {
  long i = (long)blockIdx.x * blockDim.x + threadIdx.x;
  long s = (long)gridDim.x * blockDim.x;
  for (; i < total; i += s) {
    int c = (int)((i / L) % C);
    float m = stats[c] * invcnt;
    float var = stats[C + c] * invcnt - m * m;
    float sc = gam[c] * rsqrtf(var + 1e-5f);
    float sh = bet[c] - m * sc;
    y[i] = fmaxf(x[i] * sc + sh, 0.f);
  }
}

// ---------------------------------------------------------------------------
// GCN kernels (128 features)
// ---------------------------------------------------------------------------
__global__ void k_deg(const int* __restrict__ dst, float* __restrict__ deg, long E) {
  long i = (long)blockIdx.x * blockDim.x + threadIdx.x;
  long s = (long)gridDim.x * blockDim.x;
  for (; i < E; i += s) atomicAdd(&deg[dst[i]], 1.f);
}

__global__ void k_dinv(const float* __restrict__ deg, float* __restrict__ dinv, int N) {
  int i = blockIdx.x * blockDim.x + threadIdx.x;
  if (i < N) dinv[i] = rsqrtf(fmaxf(deg[i], 1e-12f));
}

// self-loop message (plain store, initializes the aggregation buffer)
__global__ void k_gcn_self(const float* __restrict__ hlin, const float* __restrict__ dinv,
                           float* __restrict__ out, int N) {
  long total = (long)N * 128;
  long i = (long)blockIdx.x * blockDim.x + threadIdx.x;
  long s = (long)gridDim.x * blockDim.x;
  for (; i < total; i += s) {
    int n = (int)(i >> 7);
    float di = dinv[n];
    out[i] = hlin[i] * di * di;
  }
}

// per-edge scatter: 2 edges per 256-thread block, 128 features wide
__global__ void k_gcn_scatter(const float* __restrict__ hlin, const int* __restrict__ src,
                              const int* __restrict__ dst, const float* __restrict__ dinv,
                              float* __restrict__ out, long E) {
  long e = (long)blockIdx.x * 2 + (threadIdx.x >> 7);
  if (e >= E) return;
  int f = threadIdx.x & 127;
  int sN = src[e], dN = dst[e];
  float wgt = dinv[sN] * dinv[dN];
  atomicAdd(&out[(size_t)dN * 128 + f], hlin[(size_t)sN * 128 + f] * wgt);
}

__global__ void k_bias_act(const float* __restrict__ x, const float* __restrict__ bias,
                           float* __restrict__ y, long total, int C, int relu) {
  long i = (long)blockIdx.x * blockDim.x + threadIdx.x;
  long s = (long)gridDim.x * blockDim.x;
  for (; i < total; i += s) {
    float v = x[i] + bias[i % C];
    if (relu) v = fmaxf(v, 0.f);
    y[i] = v;
  }
}

// ---------------------------------------------------------------------------
// Classifier head (Nout=5, too small for WMMA)
// ---------------------------------------------------------------------------
__global__ void k_head(const float* __restrict__ x, const float* __restrict__ w,
                       const float* __restrict__ b, float* __restrict__ logits, int N) {
  long total = (long)N * 5;
  long i = (long)blockIdx.x * blockDim.x + threadIdx.x;
  if (i >= total) return;
  int n = (int)(i / 5), j = (int)(i % 5);
  float acc = b[j];
  for (int k = 0; k < 64; ++k) acc += x[(size_t)n * 64 + k] * w[j * 64 + k];
  logits[i] = acc;
}

// concat [z (128) | cls_emb[label] (32)] -> bf16 (N,160)
__global__ void k_dec_concat(const float* __restrict__ z, const int* __restrict__ labels,
                             const float* __restrict__ emb, unsigned short* __restrict__ out,
                             int N) {
  long total = (long)N * 160;
  long i = (long)blockIdx.x * blockDim.x + threadIdx.x;
  long s = (long)gridDim.x * blockDim.x;
  for (; i < total; i += s) {
    int f = (int)(i % 160);
    long n = i / 160;
    float v = (f < 128) ? z[n * 128 + f] : emb[(size_t)labels[n] * 32 + (f - 128)];
    out[i] = f2bf(v);
  }
}

// ---------------------------------------------------------------------------
// ConvTranspose1d: stride=2, pad=1, K=4; per-sample LDS-resident input
// (Cin*Lin == 2944 for all three decoder stages). Output row stride = ldy.
// ---------------------------------------------------------------------------
__global__ void k_convt(const float* __restrict__ x, const float* __restrict__ w,
                        const float* __restrict__ bias, float* __restrict__ y,
                        int Cin, int Cout, int Lin, int Lout, int ldy) {
  __shared__ float lIn[2944];
  int n = blockIdx.x;
  int nin = Cin * Lin;
  for (int i = threadIdx.x; i < nin; i += blockDim.x)
    lIn[i] = x[(size_t)n * nin + i];
  __syncthreads();
  int nout = Cout * Lout;
  for (int i = threadIdx.x; i < nout; i += blockDim.x) {
    int o = i / Lout, t = i % Lout;
    float acc = bias[o];
    // y[t] += x[l]*w[k] where 2l + k - 1 == t, k in [0,4)
    for (int k = (t & 1) ? 0 : 1; k < 4; k += 2) {
      int l = (t + 1 - k) >> 1;
      if (l >= 0 && l < Lin) {
        for (int c = 0; c < Cin; ++c)
          acc += lIn[c * Lin + l] * w[((size_t)c * Cout + o) * 4 + k];
      }
    }
    y[((size_t)n * Cout + o) * ldy + t] = acc;
  }
}

// ---------------------------------------------------------------------------
// Host launcher
// ---------------------------------------------------------------------------
extern "C" void kernel_launch(void* const* d_in, const int* in_sizes, int n_in,
                              void* d_out, int out_size, void* d_ws, size_t ws_size,
                              hipStream_t stream) {
  // ---- inputs --------------------------------------------------------------
  const float* waveforms = (const float*)d_in[0];
  const int*   edge_index = (const int*)d_in[1];
  const int*   labels    = (const int*)d_in[2];
  int pi = 3;
  const float* ec1_w = (const float*)d_in[pi++]; const float* ec1_b = (const float*)d_in[pi++];
  const float* bn1e_g = (const float*)d_in[pi++]; const float* bn1e_b = (const float*)d_in[pi++];
  const float* ec2_w = (const float*)d_in[pi++]; const float* ec2_b = (const float*)d_in[pi++];
  const float* bn2e_g = (const float*)d_in[pi++]; const float* bn2e_b = (const float*)d_in[pi++];
  const float* ec3_w = (const float*)d_in[pi++]; const float* ec3_b = (const float*)d_in[pi++];
  const float* bn3e_g = (const float*)d_in[pi++]; const float* bn3e_b = (const float*)d_in[pi++];
  const float* efc1_w = (const float*)d_in[pi++]; const float* efc1_b = (const float*)d_in[pi++];
  const float* efc2_w = (const float*)d_in[pi++]; const float* efc2_b = (const float*)d_in[pi++];
  const float* g1_w = (const float*)d_in[pi++]; const float* g1_b = (const float*)d_in[pi++];
  const float* g2_w = (const float*)d_in[pi++]; const float* g2_b = (const float*)d_in[pi++];
  const float* c1_w = (const float*)d_in[pi++]; const float* c1_b = (const float*)d_in[pi++];
  const float* c2_w = (const float*)d_in[pi++]; const float* c2_b = (const float*)d_in[pi++];
  const float* cls_emb = (const float*)d_in[pi++];
  const float* dfc1_w = (const float*)d_in[pi++]; const float* dfc1_b = (const float*)d_in[pi++];
  const float* dfc2_w = (const float*)d_in[pi++]; const float* dfc2_b = (const float*)d_in[pi++];
  const float* dc1_w = (const float*)d_in[pi++]; const float* dc1_b = (const float*)d_in[pi++];
  const float* bn1d_g = (const float*)d_in[pi++]; const float* bn1d_b = (const float*)d_in[pi++];
  const float* dc2_w = (const float*)d_in[pi++]; const float* dc2_b = (const float*)d_in[pi++];
  const float* bn2d_g = (const float*)d_in[pi++]; const float* bn2d_b = (const float*)d_in[pi++];
  const float* dc3_w = (const float*)d_in[pi++]; const float* dc3_b = (const float*)d_in[pi++];

  const int  N = in_sizes[0] / 187;
  const long E = in_sizes[1] / 2;
  const int* e_src = edge_index;
  const int* e_dst = edge_index + E;

  float* out = (float*)d_out;
  float* z_out = out;                                   // (N,128)
  float* recon_out = out + (size_t)N * 128;             // (N,1,187)
  float* logits_out = recon_out + (size_t)N * 187;      // (N,5)

  // ---- workspace carve ----------------------------------------------------
  float* ws = (float*)d_ws;
  size_t off = 0;
  auto alloc = [&](size_t nElem) { float* p = ws + off; off += (nElem + 63) & ~(size_t)63; return p; };
  float* bufA  = alloc((size_t)N * 5984);   // max conv out (N,32,187)/(N,64,93)/(N,128,46)
  float* bufB  = alloc((size_t)N * 2976);   // max pooled / (N,2944)
  float* efc1o = alloc((size_t)N * 256);
  float* feats = alloc((size_t)N * 128);
  float* hlin  = alloc((size_t)N * 128);
  float* aggb  = alloc((size_t)N * 128);
  float* hbuf  = alloc((size_t)N * 128);
  float* c1o   = alloc((size_t)N * 64);
  float* dec1o = alloc((size_t)N * 256);
  float* deg   = alloc((size_t)N);
  float* dinv  = alloc((size_t)N);
  float* stats = alloc(256);
  unsigned short* xbf0 = (unsigned short*)alloc(((size_t)N * 2944 + 1) / 2);
  unsigned short* xbf1 = (unsigned short*)alloc(((size_t)N * 2944 + 1) / 2);
  unsigned short* wbf  = (unsigned short*)alloc(850000); // 1.7M halfs
  size_t needBytes = off * sizeof(float);
  if (ws_size < needBytes) return; // insufficient workspace (deterministic no-op)

  // bf16 weight offsets (halfs)
  unsigned short* w_efc1 = wbf;                 // 256*2944
  unsigned short* w_efc2 = w_efc1 + 753664;     // 128*256
  unsigned short* w_g1   = w_efc2 + 32768;      // 128*128
  unsigned short* w_g2   = w_g1 + 16384;        // 128*128
  unsigned short* w_c1   = w_g2 + 16384;        // 64*128
  unsigned short* w_dfc1 = w_c1 + 8192;         // 256*160
  unsigned short* w_dfc2 = w_dfc1 + 40960;      // 2944*256

  auto gsblocks = [](long total) { return (unsigned)((total + 255) / 256); };
  auto cvt = [&](const float* src, unsigned short* dst, long n) {
    k_f32_to_bf16<<<gsblocks(n), 256, 0, stream>>>(src, dst, n);
  };
  auto gemm = [&](const unsigned short* A, const unsigned short* W, const float* bias,
                  float* C, unsigned short* Cbf, int M, int Nn, int K, int relu) {
    dim3 grid((M + 127) / 128, Nn / 64);
    k_gemm_bf16<<<grid, 256, 0, stream>>>(A, W, bias, C, Cbf, M, Nn, K, relu);
  };
  auto bnstats = [&](const float* x, int C, int L) {
    k_zero<<<1, 256, 0, stream>>>(stats, 256);
    k_bn_stats<<<4096, 256, 0, stream>>>(x, stats, (long)N * C * L, C, L);
  };

  // ---- init outputs & bf16 weights ---------------------------------------
  k_zero<<<gsblocks(out_size), 256, 0, stream>>>(out, (long)out_size);
  cvt(efc1_w, w_efc1, 753664);
  cvt(efc2_w, w_efc2, 32768);
  cvt(g1_w,   w_g1,   16384);
  cvt(g2_w,   w_g2,   16384);
  cvt(c1_w,   w_c1,   8192);
  cvt(dfc1_w, w_dfc1, 40960);
  cvt(dfc2_w, w_dfc2, 753664);

  // ---- ECG encoder --------------------------------------------------------
  k_conv1<<<gsblocks((long)N * 32 * 187), 256, 0, stream>>>(waveforms, ec1_w, ec1_b, bufA, N);
  bnstats(bufA, 32, 187);
  k_bn_relu_pool<<<gsblocks((long)N * 32 * 93), 256, 0, stream>>>(
      bufA, stats, bn1e_g, bn1e_b, bufB, (long)N * 32 * 93, 32, 187, 93, 1.f / ((float)N * 187));

  k_conv1d<<<N, 256, 0, stream>>>(bufB, ec2_w, ec2_b, bufA, 32, 64, 93);
  bnstats(bufA, 64, 93);
  k_bn_relu_pool<<<gsblocks((long)N * 64 * 46), 256, 0, stream>>>(
      bufA, stats, bn2e_g, bn2e_b, bufB, (long)N * 64 * 46, 64, 93, 46, 1.f / ((float)N * 93));

  k_conv1d<<<N, 256, 0, stream>>>(bufB, ec3_w, ec3_b, bufA, 64, 128, 46);
  bnstats(bufA, 128, 46);
  k_bn_relu_pool<<<gsblocks((long)N * 128 * 23), 256, 0, stream>>>(
      bufA, stats, bn3e_g, bn3e_b, bufB, (long)N * 128 * 23, 128, 46, 23, 1.f / ((float)N * 46));

  // flatten (N,2944) -> bf16, FC layers via WMMA
  cvt(bufB, xbf0, (long)N * 2944);
  gemm(xbf0, w_efc1, efc1_b, efc1o, xbf1, N, 256, 2944, 1);   // efc1 + ReLU
  gemm(xbf1, w_efc2, efc2_b, feats, nullptr, N, 128, 256, 0); // efc2 -> feats

  // ---- GCN ---------------------------------------------------------------
  k_fill<<<gsblocks(N), 256, 0, stream>>>(deg, 1.f, N);       // self-loops
  k_deg<<<gsblocks(E), 256, 0, stream>>>(e_dst, deg, E);
  k_dinv<<<(N + 255) / 256, 256, 0, stream>>>(deg, dinv, N);

  cvt(feats, xbf0, (long)N * 128);
  gemm(xbf0, w_g1, nullptr, hlin, nullptr, N, 128, 128, 0);
  k_gcn_self<<<gsblocks((long)N * 128), 256, 0, stream>>>(hlin, dinv, aggb, N);
  k_gcn_scatter<<<(unsigned)((E + 1) / 2), 256, 0, stream>>>(hlin, e_src, e_dst, dinv, aggb, E);
  k_bias_act<<<gsblocks((long)N * 128), 256, 0, stream>>>(aggb, g1_b, hbuf, (long)N * 128, 128, 1);

  cvt(hbuf, xbf0, (long)N * 128);
  gemm(xbf0, w_g2, nullptr, hlin, nullptr, N, 128, 128, 0);
  k_gcn_self<<<gsblocks((long)N * 128), 256, 0, stream>>>(hlin, dinv, aggb, N);
  k_gcn_scatter<<<(unsigned)((E + 1) / 2), 256, 0, stream>>>(hlin, e_src, e_dst, dinv, aggb, E);
  k_bias_act<<<gsblocks((long)N * 128), 256, 0, stream>>>(aggb, g2_b, z_out, (long)N * 128, 128, 0);

  // ---- Classifier ---------------------------------------------------------
  cvt(z_out, xbf0, (long)N * 128);
  gemm(xbf0, w_c1, c1_b, c1o, nullptr, N, 64, 128, 1);
  k_head<<<gsblocks((long)N * 5), 256, 0, stream>>>(c1o, c2_w, c2_b, logits_out, N);

  // ---- Class-conditioned decoder -----------------------------------------
  k_dec_concat<<<gsblocks((long)N * 160), 256, 0, stream>>>(z_out, labels, cls_emb, xbf1, N);
  gemm(xbf1, w_dfc1, dfc1_b, dec1o, xbf0, N, 256, 160, 1);    // dfc1 + ReLU -> bf16 mirror
  gemm(xbf0, w_dfc2, dfc2_b, bufB, nullptr, N, 2944, 256, 1); // dfc2 + ReLU -> (N,128,23)

  k_convt<<<N, 256, 0, stream>>>(bufB, dc1_w, dc1_b, bufA, 128, 64, 23, 46, 46);
  bnstats(bufA, 64, 46);
  k_bn_act<<<gsblocks((long)N * 64 * 46), 256, 0, stream>>>(
      bufA, stats, bn1d_g, bn1d_b, bufB, (long)N * 64 * 46, 64, 46, 1.f / ((float)N * 46));

  k_convt<<<N, 256, 0, stream>>>(bufB, dc2_w, dc2_b, bufA, 64, 32, 46, 92, 92);
  bnstats(bufA, 32, 92);
  k_bn_act<<<gsblocks((long)N * 32 * 92), 256, 0, stream>>>(
      bufA, stats, bn2d_g, bn2d_b, bufB, (long)N * 32 * 92, 32, 92, 1.f / ((float)N * 92));

  // last convT writes straight into recon (row stride 187; cols 184..186 stay 0)
  k_convt<<<N, 256, 0, stream>>>(bufB, dc3_w, dc3_b, recon_out, 32, 1, 92, 184, 187);
}
